// DecoderRNNT_15229954032087
// MI455X (gfx1250) — matrix operand order repaired and checked by reading, not compiled, not run on hardware
//
#include <hip/hip_runtime.h>
#include <hip/hip_bf16.h>

// ---------------- types for WMMA ----------------
typedef __attribute__((ext_vector_type(16))) __bf16          v16bf;
typedef __attribute__((ext_vector_type(8)))  float           v8f;
typedef __attribute__((ext_vector_type(8)))  unsigned short  ush8;

union BFrag { v16bf v; ush8 h[2]; };

__device__ __forceinline__ unsigned short f2bf(float f) {
    unsigned int u = __float_as_uint(f);
    return (unsigned short)((u + 0x7FFFu + ((u >> 16) & 1u)) >> 16); // RNE
}

__device__ __forceinline__ float sigmoidf_(float x) {
    return 1.0f / (1.0f + __expf(-x));
}

// A-matrix (MxK=16x32, bf16) fragment in this code's contiguous k order (not the ISA table order):
// lanes 0-15 & 16-31 both map rows M=0..15; lane<16 holds K {0..15},
// lane>=16 holds K {16..31} (contiguous, matches B). Two 16B loads per lane.
__device__ __forceinline__ v16bf load_a_frag(const unsigned short* base, int stride,
                                             int row, int k0, int lane) {
    const unsigned short* p = base + (size_t)row * stride + k0 + ((lane >= 16) ? 16 : 0);
    BFrag f;
    f.h[0] = *(const ush8*)(p);
    f.h[1] = *(const ush8*)(p + 8);
    return f.v;
}

// B-matrix (KxN=32x16, bf16) fragment: lane l -> column N=l%16,
// lane<16 holds K=k0..k0+15, lane>=16 holds K=k0+16..k0+31 (contiguous).
// W is stored row-major (N rows, K cols) = B transposed => one 32B load.
__device__ __forceinline__ v16bf load_b_frag(const unsigned short* base, int stride,
                                             int nrow, int k0, int lane) {
    const unsigned short* p = base + (size_t)nrow * stride + k0 + ((lane >= 16) ? 16 : 0);
    return *(const v16bf*)(p);
}

#define WMMA_BF16(a, b, c) \
    __builtin_amdgcn_wmma_f32_16x16x32_bf16(false, (a), false, (b), (short)0, (c), false, false)

// ---------------- small prep kernels ----------------
__global__ void k_f32_to_bf16(unsigned short* __restrict__ dst,
                              const float* __restrict__ src, int n) {
    int i = blockIdx.x * blockDim.x + threadIdx.x;
    if (i < n) dst[i] = f2bf(src[i]);
}

// lin_out_w (500x512) -> padded (512x512) bf16, rows 500..511 = 0
__global__ void k_pad_outw(unsigned short* __restrict__ dst,
                           const float* __restrict__ src) {
    int i = blockIdx.x * blockDim.x + threadIdx.x;
    if (i < 512 * 512) {
        int row = i >> 9;
        dst[i] = (row < 500) ? f2bf(src[i]) : (unsigned short)0;
    }
}

__global__ void k_vadd(float* __restrict__ dst, const float* __restrict__ a,
                       const float* __restrict__ b, int n) {
    int i = blockIdx.x * blockDim.x + threadIdx.x;
    if (i < n) dst[i] = a[i] + b[i];
}

// eys[b,u,:] = bf16(embed[ys[b,u],:]) ; one block per (b,u)
__global__ void k_embed(unsigned short* __restrict__ eys,
                        const float* __restrict__ embed,
                        const int* __restrict__ ys) {
    int bu = blockIdx.x;                 // 0..639  (= b*80+u)
    int tok = ys[bu];
    const float* row = embed + (size_t)tok * 512;
    unsigned short* dst = eys + (size_t)bu * 512;
    for (int j = threadIdx.x; j < 512; j += blockDim.x) dst[j] = f2bf(row[j]);
}

// ---------------- generic bf16 WMMA GEMM: C(MxN,f32) = A(MxK) @ W(NxK)^T + bias ----------------
// grid(M/16, N/64), block 128 (4 waves; wave w -> 16-wide N subtile)
__global__ void __launch_bounds__(128) k_gemm(
    float* __restrict__ C, const unsigned short* __restrict__ A,
    const unsigned short* __restrict__ W, const float* __restrict__ bias,
    int M, int N, int K) {
    int lane = threadIdx.x & 31;
    int wave = threadIdx.x >> 5;
    int m0 = blockIdx.x * 16;
    int n0 = blockIdx.y * 64 + wave * 16;
    int arow = m0 + (lane & 15);
    int nrow = n0 + (lane & 15);
    v8f acc = {};
    for (int k0 = 0; k0 < K; k0 += 32) {
        v16bf a = load_a_frag(A, K, arow, k0, lane);
        v16bf b = load_b_frag(W, K, nrow, k0, lane);
        acc = WMMA_BF16(a, b, acc);
    }
    int mbase = m0 + ((lane >= 16) ? 8 : 0);
    int n = n0 + (lane & 15);
    float bv = bias ? bias[n] : 0.0f;
#pragma unroll
    for (int r = 0; r < 8; ++r)
        C[(size_t)(mbase + r) * N + n] = acc[r] + bv;
}

// ---------------- one LSTM cell step (one layer) ----------------
// grid 32 (j-tile of 16 hidden units), block 128 (wave w = gate w: i,f,g,o)
// gates = hprev @ Whh^T [+ xin @ Wih^T] + (X0 row | bias)
__global__ void __launch_bounds__(128) k_lstm_cell(
    const unsigned short* __restrict__ hprev,   // 16x512 bf16 (rows 8..15 zero)
    const unsigned short* __restrict__ Whh,     // 2048x512 bf16
    const unsigned short* __restrict__ xin,     // 16x512 bf16 or null
    const unsigned short* __restrict__ Wih,     // 2048x512 bf16 or null
    const float* __restrict__ X0,               // null or (640x2048), rows b*80+u (incl. biases)
    const float* __restrict__ bias,             // null or [2048]
    float* __restrict__ c,                      // 8x512 f32 (in/out, per-j slice)
    unsigned short* __restrict__ hout,          // 16x512 bf16
    unsigned short* __restrict__ hdec,          // null or (8*80 x 512) bf16
    int t) {
    __shared__ float gsh[4][8][16];
    int lane = threadIdx.x & 31;
    int wave = threadIdx.x >> 5;
    int j0 = blockIdx.x * 16;
    int n0 = wave * 512 + j0;                   // gate `wave`, units j0..j0+15
    int arow = lane & 15;
    int nrow = n0 + (lane & 15);
    v8f acc = {};
    for (int k0 = 0; k0 < 512; k0 += 32) {
        v16bf a = load_a_frag(hprev, 512, arow, k0, lane);
        v16bf b = load_b_frag(Whh, 512, nrow, k0, lane);
        acc = WMMA_BF16(a, b, acc);
    }
    if (Wih) {
        for (int k0 = 0; k0 < 512; k0 += 32) {
            v16bf a = load_a_frag(xin, 512, arow, k0, lane);
            v16bf b = load_b_frag(Wih, 512, nrow, k0, lane);
            acc = WMMA_BF16(a, b, acc);
        }
    }
    int mbase = (lane >= 16) ? 8 : 0;
    int n = n0 + (lane & 15);
#pragma unroll
    for (int r = 0; r < 8; ++r) {
        int m = mbase + r;                      // batch row
        if (m < 8) {
            float g = acc[r];
            g += X0 ? X0[((size_t)m * 80 + t) * 2048 + n] : bias[n];
            gsh[wave][m][lane & 15] = g;
        }
    }
    __syncthreads();
    if (threadIdx.x < 128) {
        int bb = threadIdx.x >> 4;
        int j  = threadIdx.x & 15;
        float gi = gsh[0][bb][j], gf = gsh[1][bb][j];
        float gg = gsh[2][bb][j], go = gsh[3][bb][j];
        int cidx = bb * 512 + j0 + j;
        float cn = sigmoidf_(gf) * c[cidx] + sigmoidf_(gi) * tanhf(gg);
        c[cidx] = cn;
        float h = sigmoidf_(go) * tanhf(cn);
        unsigned short hb = f2bf(h);
        hout[cidx] = hb;
        if (hdec) hdec[((size_t)bb * 80 + t) * 512 + j0 + j] = hb;
    }
}

// ---------------- fused joint kernel ----------------
// 1 WG per (b,t). z(16x512) tile built in LDS from tanh(enc+dec), then
// out[b,t,u,:500] = z @ Wout^T + bias via WMMA. 8 waves x 64-wide N slices.
__global__ void __launch_bounds__(256) k_joint(
    float* __restrict__ out, const float* __restrict__ encp,
    const float* __restrict__ decp, const unsigned short* __restrict__ Wout,
    const float* __restrict__ bout) {
    __shared__ float enc_s[512];
    __shared__ unsigned short zlds[16 * 512];
    int bt = blockIdx.x;                        // b*160+t
    int bb = bt / 160;
    int tid = threadIdx.x;
    int lane = tid & 31;
    int wave = tid >> 5;
    const float* encrow = encp + (size_t)bt * 512;
    for (int i = tid; i < 512; i += 256) enc_s[i] = encrow[i];
    __syncthreads();

    for (int mt = 0; mt < 5; ++mt) {
        int u0 = mt * 16;
        const float* decbase = decp + ((size_t)bb * 80 + u0) * 512;
        // build z tile: z[uu][j] = tanh(enc[j] + dec[u0+uu][j]), bf16
#pragma unroll 4
        for (int i = 0; i < 32; ++i) {
            int e = tid + i * 256;              // 0..8191
            int uu = e >> 9;
            int j  = e & 511;
            float zv = tanhf(enc_s[j] + decbase[(size_t)uu * 512 + j]);
            zlds[e] = f2bf(zv);
        }
        __syncthreads();

        int n0 = wave * 64;
        v8f acc[4] = {v8f{}, v8f{}, v8f{}, v8f{}};
        int arow = lane & 15;
        int ncol = lane & 15;
        for (int k0 = 0; k0 < 512; k0 += 32) {
            v16bf a  = load_a_frag((const unsigned short*)zlds, 512, arow, k0, lane);
            v16bf w0 = load_b_frag(Wout, 512, n0 +  0 + ncol, k0, lane);
            v16bf w1 = load_b_frag(Wout, 512, n0 + 16 + ncol, k0, lane);
            v16bf w2 = load_b_frag(Wout, 512, n0 + 32 + ncol, k0, lane);
            v16bf w3 = load_b_frag(Wout, 512, n0 + 48 + ncol, k0, lane);
            acc[0] = WMMA_BF16(a, w0, acc[0]);
            acc[1] = WMMA_BF16(a, w1, acc[1]);
            acc[2] = WMMA_BF16(a, w2, acc[2]);
            acc[3] = WMMA_BF16(a, w3, acc[3]);
        }
        __syncthreads();                        // zlds free for next tile

        size_t outbase = ((size_t)bt * 80 + u0) * 500;
        int mb = (lane >= 16) ? 8 : 0;
#pragma unroll
        for (int sub = 0; sub < 4; ++sub) {
            int v = n0 + sub * 16 + ncol;
            if (v < 500) {
                float bv = bout[v];
#pragma unroll
                for (int r = 0; r < 8; ++r) {
                    int u = u0 + mb + r;
                    out[outbase - (size_t)u0 * 500 + (size_t)u * 500 + v] = acc[sub][r] + bv;
                }
            }
        }
    }
}

// ---------------- host-side orchestration ----------------
extern "C" void kernel_launch(void* const* d_in, const int* in_sizes, int n_in,
                              void* d_out, int out_size, void* d_ws, size_t ws_size,
                              hipStream_t stream) {
    (void)in_sizes; (void)n_in; (void)out_size; (void)ws_size;
    const float* hs_pad    = (const float*)d_in[0];
    const int*   ys        = (const int*)  d_in[1];
    const float* embed     = (const float*)d_in[2];
    const float* W_ih0     = (const float*)d_in[3];
    const float* W_hh0     = (const float*)d_in[4];
    const float* b_ih0     = (const float*)d_in[5];
    const float* b_hh0     = (const float*)d_in[6];
    const float* W_ih1     = (const float*)d_in[7];
    const float* W_hh1     = (const float*)d_in[8];
    const float* b_ih1     = (const float*)d_in[9];
    const float* b_hh1     = (const float*)d_in[10];
    const float* lin_enc_w = (const float*)d_in[11];
    const float* lin_enc_b = (const float*)d_in[12];
    const float* lin_dec_w = (const float*)d_in[13];
    const float* lin_out_w = (const float*)d_in[14];
    const float* lin_out_b = (const float*)d_in[15];
    float* out = (float*)d_out;

    char* ws = (char*)d_ws;
    size_t off = 0;
    auto alloc = [&](size_t bytes) -> void* {
        void* p = ws + off;
        off += (bytes + 255) & ~(size_t)255;
        return p;
    };
    unsigned short* hs_bf   = (unsigned short*)alloc(2ull * 8 * 160 * 512);
    unsigned short* eys_bf  = (unsigned short*)alloc(2ull * 8 * 80 * 512);
    unsigned short* Wih0_bf = (unsigned short*)alloc(2ull * 2048 * 512);
    unsigned short* Whh0_bf = (unsigned short*)alloc(2ull * 2048 * 512);
    unsigned short* Wih1_bf = (unsigned short*)alloc(2ull * 2048 * 512);
    unsigned short* Whh1_bf = (unsigned short*)alloc(2ull * 2048 * 512);
    unsigned short* encw_bf = (unsigned short*)alloc(2ull * 512 * 512);
    unsigned short* decw_bf = (unsigned short*)alloc(2ull * 512 * 512);
    unsigned short* outw_bf = (unsigned short*)alloc(2ull * 512 * 512);
    float*          bsum0   = (float*)alloc(4ull * 2048);
    float*          bsum1   = (float*)alloc(4ull * 2048);
    float*          X0      = (float*)alloc(4ull * 640 * 2048);
    unsigned short* h0A     = (unsigned short*)alloc(2ull * 16 * 512);
    unsigned short* h0B     = (unsigned short*)alloc(2ull * 16 * 512);
    unsigned short* h1A     = (unsigned short*)alloc(2ull * 16 * 512);
    unsigned short* h1B     = (unsigned short*)alloc(2ull * 16 * 512);
    float*          c0      = (float*)alloc(4ull * 8 * 512);
    float*          c1      = (float*)alloc(4ull * 8 * 512);
    unsigned short* hdec_bf = (unsigned short*)alloc(2ull * 8 * 80 * 512);
    float*          encp    = (float*)alloc(4ull * 8 * 160 * 512);
    float*          decp    = (float*)alloc(4ull * 8 * 80 * 512);

    dim3 blk(256);
    auto g1 = [](int n) { return dim3((unsigned)((n + 255) / 256)); };

    // bf16 conversions
    k_f32_to_bf16<<<g1(8 * 160 * 512), blk, 0, stream>>>(hs_bf, hs_pad, 8 * 160 * 512);
    k_f32_to_bf16<<<g1(2048 * 512), blk, 0, stream>>>(Wih0_bf, W_ih0, 2048 * 512);
    k_f32_to_bf16<<<g1(2048 * 512), blk, 0, stream>>>(Whh0_bf, W_hh0, 2048 * 512);
    k_f32_to_bf16<<<g1(2048 * 512), blk, 0, stream>>>(Wih1_bf, W_ih1, 2048 * 512);
    k_f32_to_bf16<<<g1(2048 * 512), blk, 0, stream>>>(Whh1_bf, W_hh1, 2048 * 512);
    k_f32_to_bf16<<<g1(512 * 512), blk, 0, stream>>>(encw_bf, lin_enc_w, 512 * 512);
    k_f32_to_bf16<<<g1(512 * 512), blk, 0, stream>>>(decw_bf, lin_dec_w, 512 * 512);
    k_pad_outw<<<g1(512 * 512), blk, 0, stream>>>(outw_bf, lin_out_w);
    k_vadd<<<g1(2048), blk, 0, stream>>>(bsum0, b_ih0, b_hh0, 2048);
    k_vadd<<<g1(2048), blk, 0, stream>>>(bsum1, b_ih1, b_hh1, 2048);
    k_embed<<<dim3(640), dim3(128), 0, stream>>>(eys_bf, embed, ys);

    // zero LSTM state (h0A..c1 are contiguous: 4x16KB bf16 + 2x16KB f32)
    hipMemsetAsync(h0A, 0, 6 * 16384, stream);

    // X0 = eys @ W_ih0^T + (b_ih0 + b_hh0)   (640 x 2048)
    k_gemm<<<dim3(40, 32), dim3(128), 0, stream>>>(X0, eys_bf, Wih0_bf, bsum0, 640, 2048, 512);

    // sequential 2-layer LSTM, ping-pong h buffers
    for (int t = 0; t < 80; ++t) {
        unsigned short* h0p = (t & 1) ? h0B : h0A;
        unsigned short* h0n = (t & 1) ? h0A : h0B;
        unsigned short* h1p = (t & 1) ? h1B : h1A;
        unsigned short* h1n = (t & 1) ? h1A : h1B;
        k_lstm_cell<<<dim3(32), dim3(128), 0, stream>>>(
            h0p, Whh0_bf, nullptr, nullptr, X0, nullptr, c0, h0n, nullptr, t);
        k_lstm_cell<<<dim3(32), dim3(128), 0, stream>>>(
            h1p, Whh1_bf, h0n, Wih1_bf, nullptr, bsum1, c1, h1n, hdec_bf, t);
    }

    // projections
    k_gemm<<<dim3(80, 8), dim3(128), 0, stream>>>(encp, hs_bf, encw_bf, lin_enc_b, 1280, 512, 512);
    k_gemm<<<dim3(40, 8), dim3(128), 0, stream>>>(decp, hdec_bf, decw_bf, nullptr, 640, 512, 512);

    // fused joint
    k_joint<<<dim3(1280), dim3(256), 0, stream>>>(out, encp, decp, outw_bf, lin_out_b);
}
